// GVPBackbone_94489280930
// MI455X (gfx1250) — matrix operand order repaired and checked by reading, not compiled, not run on hardware
//
// GVP-GNN backbone forward for MI455X (gfx1250), wave32 + V_WMMA_F32_16X16X32_BF16.
//
// Strategy (compile-only reasoning from MI455X specs):
//  * All node state (s: 8.4MB, v: 3.1MB) + accumulators fit in the 192MB L2, so
//    random edge gathers and f32 atomics for mean-aggregation stay on-chip.
//  * Every dense GEMM (message ws 321x128 / 144x128, ff 160x512 / 544x128,
//    gates so x vo, head 128x128) is tiled M=16 rows per wave32 and computed
//    with v_wmma_f32_16x16x32_bf16 (f32 accumulate). Weights are re-packed each
//    launch into a fragment-ready bf16 layout (32B contiguous per lane/tile).
//  * vh/wv vector-channel mixing, norms and sigmoid gates (<15% of FLOPs) on VALU.
//
// Input pointer order (top-level dict insertion order; nested params flattened
// in jax pytree (sorted-key) order):
//  0 x_s [N,6]  1 x_v [N,3,3]  2 edge_s [E,32]  3 edge_v [E,1,3]
//  4 edge_index [2,E] i32      5 batch [N] i32
//  6 head.b1  7 head.b2  8 head.w1 [128,128]  9 head.w2 [128,1]
//  10..15 input_proj {wh, ws_b, ws_w, wsv_b, wsv_w, wv}
//  16+34*l .. : layer l = ff0{6}, ff1{6}, ln0{beta,gamma}, ln1{beta,gamma},
//               msg0{6}, msg1{6}, msg2{6}   (each GVP: wh, ws_b, ws_w, wsv_b, wsv_w, wv)

#include <hip/hip_runtime.h>
#include <stdint.h>

typedef __attribute__((ext_vector_type(16))) __bf16 v16bf;
typedef __attribute__((ext_vector_type(8)))  float  v8f;

#define EPSF 1e-8f

__device__ __forceinline__ unsigned short f2bf(float x) {
  unsigned u = __float_as_uint(x);
  u += 0x7FFFu + ((u >> 16) & 1u);          // round-to-nearest-even
  return (unsigned short)(u >> 16);
}
__device__ __forceinline__ unsigned pack2bf(float a, float b) {
  return (unsigned)f2bf(a) | ((unsigned)f2bf(b) << 16);
}
__device__ __forceinline__ float sigmoidf(float x) {
  return 1.0f / (1.0f + __expf(-x));
}

union AFrag { v16bf v; unsigned u[8]; };
union BFrag { v16bf v; uint4   q[2]; };

// ---------------------------------------------------------------------------
// Wave-scope GEMM: C[16 x 16*NT] = A[16 x 32*KT] * Wpacked + bias  (f32 out)
// A source: amode 0 = bf16 LDS (row-major, stride Ast elements)
//           amode 1 = f32  LDS (convert on the fly)
//           amode 2 = f32  LDS with relu before convert
// Wpacked: fragment-ready bf16, tile (kt*NT+nt): 512 el, lane*16 contiguous.
// ---------------------------------------------------------------------------
__device__ __forceinline__ void wave_gemm(
    const unsigned short* Abf, const float* Af32, int amode, int Ast, int KT,
    const unsigned short* Wp, int NT, const float* bias,
    float* outS, int ost, int lane)
{
  const int r    = lane & 15;
  const int koff = (lane >> 4) * 8;
  for (int nt = 0; nt < NT; ++nt) {
    v8f c = {0.f, 0.f, 0.f, 0.f, 0.f, 0.f, 0.f, 0.f};
    for (int kt = 0; kt < KT; ++kt) {
      AFrag a;
      const int kb = kt * 32 + koff;
#pragma unroll
      for (int j = 0; j < 8; ++j) {
        const int k = kb + ((j < 4) ? (2 * j) : (16 + 2 * (j - 4)));
        if (amode == 0) {
          a.u[j] = *(const unsigned*)(Abf + (size_t)r * Ast + k);
        } else {
          float x0 = Af32[(size_t)r * Ast + k];
          float x1 = Af32[(size_t)r * Ast + k + 1];
          if (amode == 2) { x0 = fmaxf(x0, 0.f); x1 = fmaxf(x1, 0.f); }
          a.u[j] = pack2bf(x0, x1);
        }
      }
      BFrag b;
      const uint4* bp = (const uint4*)(Wp + ((size_t)(kt * NT + nt)) * 512 + lane * 16);
      b.q[0] = bp[0];
      b.q[1] = bp[1];
      c = __builtin_amdgcn_wmma_f32_16x16x32_bf16(false, a.v, false, b.v,
                                                  (short)0, c, false, false);
    }
#pragma unroll
    for (int g = 0; g < 8; ++g) {
      const int row = g + (lane >> 4) * 8;
      const int col = nt * 16 + (lane & 15);
      outS[(size_t)row * ost + col] = c[g] + bias[col];
    }
  }
}

// ---------------------------------------------------------------------------
// Weight packer: W[K x N] f32 row-major -> fragment-ready bf16 tiles.
// ---------------------------------------------------------------------------
__global__ __launch_bounds__(256) void pack_kernel(
    const float* __restrict__ W, int K, int N, int NT,
    unsigned short* __restrict__ dst)
{
  const int tile = blockIdx.x;
  const int kt = tile / NT, nt = tile % NT;
  for (int t = threadIdx.x; t < 512; t += 256) {
    const int ln = t >> 4;
    const int j  = (t >> 1) & 7;
    const int h  = t & 1;
    const int n  = nt * 16 + (ln & 15);
    const int k  = kt * 32 + (j >> 2) * 16 + (ln >> 4) * 8 + 2 * (j & 3) + h;
    unsigned short val = 0;
    if (k < K && n < N) val = f2bf(W[(size_t)k * N + n]);
    dst[(size_t)tile * 512 + t] = val;
  }
}

__global__ void zero_kernel(float* p, long n) {
  long t = (long)blockIdx.x * blockDim.x + threadIdx.x;
  if (t < n) p[t] = 0.f;
}

// ---------------------------------------------------------------------------
// Input projection GVP (si=6, vi=3 -> so=128, vo=16, h=16), VALU only.
// One block (128 thr) per node.
// ---------------------------------------------------------------------------
__global__ __launch_bounds__(128) void input_proj_kernel(
    const float* __restrict__ xs, const float* __restrict__ xv,
    const float* __restrict__ wh, const float* __restrict__ wsw,
    const float* __restrict__ wsb, const float* __restrict__ wsvw,
    const float* __restrict__ wsvb, const float* __restrict__ wv,
    float* __restrict__ s, float* __restrict__ v)
{
  __shared__ float vh[48];
  __shared__ float vn[16];
  __shared__ float sr[128];
  __shared__ float gt[16];
  const int n = blockIdx.x, t = threadIdx.x;
  if (t < 48) {
    const int h = t / 3, c = t % 3;
    float acc = 0.f;
#pragma unroll
    for (int i = 0; i < 3; ++i) acc += xv[(size_t)n * 9 + i * 3 + c] * wh[i * 16 + h];
    vh[t] = acc;
  }
  __syncthreads();
  if (t < 16) {
    float nn = 0.f;
#pragma unroll
    for (int c = 0; c < 3; ++c) { const float x = vh[t * 3 + c]; nn += x * x; }
    vn[t] = sqrtf(fmaxf(nn, EPSF));
  }
  __syncthreads();
  {
    float acc = wsb[t];
#pragma unroll
    for (int k = 0; k < 6; ++k) acc += xs[(size_t)n * 6 + k] * wsw[k * 128 + t];
#pragma unroll
    for (int h = 0; h < 16; ++h) acc += vn[h] * wsw[(6 + h) * 128 + t];
    sr[t] = acc;
  }
  __syncthreads();
  if (t < 16) {
    float acc = wsvb[t];
    for (int c = 0; c < 128; ++c) acc += sr[c] * wsvw[c * 16 + t];
    gt[t] = sigmoidf(acc);
  }
  __syncthreads();
  if (t < 48) {
    const int o = t / 3, c = t % 3;
    float acc = 0.f;
#pragma unroll
    for (int h = 0; h < 16; ++h) acc += vh[h * 3 + c] * wv[h * 16 + o];
    v[(size_t)n * 48 + t] = acc * gt[o];
  }
  s[(size_t)n * 128 + t] = fmaxf(sr[t], 0.f);
}

// ---------------------------------------------------------------------------
// Edge-message kernel: 4 waves/block, one 16-edge tile per wave.
// 3 chained GVPs; ws / wsv matmuls on WMMA, vh/wv/norm on VALU.
// ---------------------------------------------------------------------------
#define MSG_ARENA 36352
__global__ __launch_bounds__(128) void msg_kernel(
    const int* __restrict__ eidx, const float* __restrict__ es,
    const float* __restrict__ ev,
    const float* __restrict__ s, const float* __restrict__ v,
    float* __restrict__ sacc, float* __restrict__ vacc, float* __restrict__ cnt,
    const float* wh0, const float* wv0, const float* wsb0, const float* wsvb0,
    const float* wh1, const float* wv1, const float* wsb1, const float* wsvb1,
    const float* wh2, const float* wv2, const float* wsb2, const float* wsvb2,
    const unsigned short* pm0ws, const unsigned short* pm0wsv,
    const unsigned short* pm1ws, const unsigned short* pm1wsv,
    const unsigned short* pm2ws, const unsigned short* pm2wsv,
    int E)
{
  extern __shared__ char smem[];
  const int wave = threadIdx.x >> 5;
  const int lane = threadIdx.x & 31;
  char* arena = smem + (size_t)wave * MSG_ARENA;
  unsigned short* Abf = (unsigned short*)arena;          // [16][352] bf16
  float* sraw = (float*)(arena + 11264);                 // [16][128]
  float* mv   = (float*)(arena + 19456);                 // [16][33][3]
  float* vh   = (float*)(arena + 25792);                 // [16][33][3]
  float* vcur = (float*)(arena + 32128);                 // [16][16][3]
  float* gate = (float*)(arena + 35200);                 // [16][16]
  int*   idx  = (int*)(arena + 36224);                   // [32]

  const long e0 = ((long)blockIdx.x * 4 + wave) * 16;
  if (lane < 16) {
    idx[lane]      = eidx[e0 + lane];               // src
    idx[16 + lane] = eidx[(size_t)E + e0 + lane];   // dst
  }
  __syncthreads();

  // ---- gather: ms = [s_src | e_s | s_dst], mv = [v_src | e_v | v_dst] ----
  for (int t = lane; t < 16 * 128; t += 32) {
    const int e = t >> 7, c = t & 127;
    Abf[e * 352 + c]       = f2bf(s[(size_t)idx[e] * 128 + c]);
    Abf[e * 352 + 160 + c] = f2bf(s[(size_t)idx[16 + e] * 128 + c]);
  }
  for (int t = lane; t < 16 * 32; t += 32) {
    const int e = t >> 5, c = t & 31;
    Abf[e * 352 + 128 + c] = f2bf(es[(e0 + e) * 32 + c]);
  }
  for (int t = lane; t < 16 * 16; t += 32) {
    const int e = t >> 4, i = t & 15;
#pragma unroll
    for (int c = 0; c < 3; ++c) {
      mv[e * 99 + i * 3 + c]        = v[(size_t)idx[e] * 48 + i * 3 + c];
      mv[e * 99 + (17 + i) * 3 + c] = v[(size_t)idx[16 + e] * 48 + i * 3 + c];
    }
  }
  for (int t = lane; t < 16; t += 32) {
#pragma unroll
    for (int c = 0; c < 3; ++c) mv[t * 99 + 48 + c] = ev[(e0 + t) * 3 + c];
  }
  __syncthreads();

  // ---- stage 1: vh = mv @ wh0 (33x33) ----
  for (int t = lane; t < 16 * 99; t += 32) {
    const int e = t / 99, rem = t % 99, h = rem / 3, c = rem % 3;
    float acc = 0.f;
    for (int i = 0; i < 33; ++i) acc += mv[e * 99 + i * 3 + c] * wh0[i * 33 + h];
    vh[e * 99 + h * 3 + c] = acc;
  }
  __syncthreads();
  for (int t = lane; t < 16 * 33; t += 32) {
    const int e = t / 33, h = t % 33;
    float nn = 0.f;
#pragma unroll
    for (int c = 0; c < 3; ++c) { const float x = vh[e * 99 + h * 3 + c]; nn += x * x; }
    Abf[e * 352 + 288 + h] = f2bf(sqrtf(fmaxf(nn, EPSF)));
  }
  for (int t = lane; t < 16 * 31; t += 32) {   // zero pad cols 321..351
    Abf[(t / 31) * 352 + 321 + (t % 31)] = 0;
  }
  __syncthreads();
  wave_gemm(Abf, nullptr, 0, 352, 11, pm0ws, 8, wsb0, sraw, 128, lane);
  __syncthreads();
  wave_gemm(nullptr, sraw, 1, 128, 4, pm0wsv, 1, wsvb0, gate, 16, lane);
  __syncthreads();
  for (int t = lane; t < 16 * 48; t += 32) {
    const int e = t / 48, rem = t % 48, o = rem / 3, c = rem % 3;
    float acc = 0.f;
    for (int h = 0; h < 33; ++h) acc += vh[e * 99 + h * 3 + c] * wv0[h * 16 + o];
    vcur[e * 48 + o * 3 + c] = acc * sigmoidf(gate[e * 16 + o]);
  }
  __syncthreads();

  // ---- stages 2 & 3 (si=128, vi=16, so=128, vo=16, h=16) ----
  for (int st = 0; st < 2; ++st) {
    const float* whx   = st ? wh2 : wh1;
    const float* wvx   = st ? wv2 : wv1;
    const float* wsbx  = st ? wsb2 : wsb1;
    const float* wsvbx = st ? wsvb2 : wsvb1;
    const unsigned short* pws  = st ? pm2ws  : pm1ws;
    const unsigned short* pwsv = st ? pm2wsv : pm1wsv;

    for (int t = lane; t < 16 * 128; t += 32) {      // A cols 0..127 = relu(s_prev)
      const int e = t >> 7, c = t & 127;
      Abf[e * 352 + c] = f2bf(fmaxf(sraw[e * 128 + c], 0.f));
    }
    for (int t = lane; t < 16 * 48; t += 32) {       // vh = vcur @ wh (16x16)
      const int e = t / 48, rem = t % 48, h = rem / 3, c = rem % 3;
      float acc = 0.f;
#pragma unroll
      for (int i = 0; i < 16; ++i) acc += vcur[e * 48 + i * 3 + c] * whx[i * 16 + h];
      vh[e * 99 + h * 3 + c] = acc;
    }
    __syncthreads();
    for (int t = lane; t < 16 * 16; t += 32) {
      const int e = t >> 4, h = t & 15;
      float nn = 0.f;
#pragma unroll
      for (int c = 0; c < 3; ++c) { const float x = vh[e * 99 + h * 3 + c]; nn += x * x; }
      Abf[e * 352 + 128 + h] = f2bf(sqrtf(fmaxf(nn, EPSF)));
    }
    for (int t = lane; t < 16 * 16; t += 32)         // zero pad cols 144..159
      Abf[(t >> 4) * 352 + 144 + (t & 15)] = 0;
    __syncthreads();
    wave_gemm(Abf, nullptr, 0, 352, 5, pws, 8, wsbx, sraw, 128, lane);
    __syncthreads();
    wave_gemm(nullptr, sraw, 1, 128, 4, pwsv, 1, wsvbx, gate, 16, lane);
    __syncthreads();
    for (int t = lane; t < 16 * 48; t += 32) {
      const int e = t / 48, rem = t % 48, o = rem / 3, c = rem % 3;
      float acc = 0.f;
#pragma unroll
      for (int h = 0; h < 16; ++h) acc += vh[e * 99 + h * 3 + c] * wvx[h * 16 + o];
      vcur[e * 48 + o * 3 + c] = acc * sigmoidf(gate[e * 16 + o]);
    }
    __syncthreads();
  }

  // ---- mean-aggregation: atomics into L2-resident accumulators ----
  for (int t = lane; t < 16 * 128; t += 32) {
    const int e = t >> 7, c = t & 127;
    atomicAdd(&sacc[(size_t)idx[16 + e] * 128 + c], sraw[e * 128 + c]);
  }
  for (int t = lane; t < 16 * 48; t += 32) {
    const int e = t / 48;
    atomicAdd(&vacc[(size_t)idx[16 + e] * 48 + (t % 48)], vcur[e * 48 + (t % 48)]);
  }
  if (lane < 16) atomicAdd(&cnt[idx[16 + lane]], 1.0f);
}

// ---------------------------------------------------------------------------
// Node update: s,v <- LayerNorm0(s + dh_s/cnt, v + dh_v/cnt). One block/node.
// ---------------------------------------------------------------------------
__global__ __launch_bounds__(128) void node_update_kernel(
    float* __restrict__ s, float* __restrict__ v,
    const float* __restrict__ sacc, const float* __restrict__ vacc,
    const float* __restrict__ cnt,
    const float* __restrict__ gamma, const float* __restrict__ beta)
{
  __shared__ float ns[128];
  __shared__ float nv[48];
  __shared__ float st[3];   // mu, rstd, vnorm
  const int n = blockIdx.x, t = threadIdx.x;
  const float cn = fmaxf(cnt[n], 1.f);
  ns[t] = s[(size_t)n * 128 + t] + sacc[(size_t)n * 128 + t] / cn;
  if (t < 48) nv[t] = v[(size_t)n * 48 + t] + vacc[(size_t)n * 48 + t] / cn;
  __syncthreads();
  if (t == 0) {
    float mu = 0.f;
    for (int c = 0; c < 128; ++c) mu += ns[c];
    mu *= (1.f / 128.f);
    float var = 0.f;
    for (int c = 0; c < 128; ++c) { const float d = ns[c] - mu; var += d * d; }
    st[0] = mu;
    st[1] = rsqrtf(var * (1.f / 128.f) + 1e-5f);
  } else if (t == 1) {
    float ss = 0.f;
    for (int i = 0; i < 16; ++i) {
      float q = 0.f;
#pragma unroll
      for (int c = 0; c < 3; ++c) { const float x = nv[i * 3 + c]; q += x * x; }
      ss += fmaxf(q, EPSF);
    }
    st[2] = sqrtf(ss * (1.f / 16.f));
  }
  __syncthreads();
  s[(size_t)n * 128 + t] = (ns[t] - st[0]) * st[1] * gamma[t] + beta[t];
  if (t < 48) v[(size_t)n * 48 + t] = nv[t] / st[2];
}

// ---------------------------------------------------------------------------
// Fused feed-forward (2 GVPs) + residual + LayerNorm1.
// 2 waves/block, one 16-node tile per wave.
// ---------------------------------------------------------------------------
#define FF_ARENA 75776
__global__ __launch_bounds__(64) void ff_kernel(
    float* __restrict__ s, float* __restrict__ v,
    const float* wh0, const float* wv0, const float* wsb0, const float* wsvb0,
    const float* wh1, const float* wv1, const float* wsb1, const float* wsvb1,
    const unsigned short* pf0ws, const unsigned short* pf0wsv,
    const unsigned short* pf1ws, const unsigned short* pf1wsv,
    const float* __restrict__ gamma, const float* __restrict__ beta)
{
  extern __shared__ char smem[];
  const int wave = threadIdx.x >> 5;
  const int lane = threadIdx.x & 31;
  char* arena = smem + (size_t)wave * FF_ARENA;
  unsigned short* Abf = (unsigned short*)arena;      // [16][544] bf16
  float* s1   = (float*)(arena + 17408);             // [16][512]
  float* s2   = (float*)(arena + 50176);             // [16][128]
  float* vin  = (float*)(arena + 58368);             // [16][16][3]
  float* vb   = (float*)(arena + 61440);             // [16][32][3]
  float* vhb  = (float*)(arena + 67584);             // [16][32][3]
  float* gate = (float*)(arena + 73728);             // [16][32]

  const int n0 = (blockIdx.x * 2 + wave) * 16;

  for (int t = lane; t < 16 * 48; t += 32)
    vin[t] = v[(size_t)(n0 + t / 48) * 48 + (t % 48)];
  for (int t = lane; t < 16 * 128; t += 32) {
    const int e = t >> 7, c = t & 127;
    Abf[e * 544 + c] = f2bf(s[(size_t)(n0 + e) * 128 + c]);
  }
  __syncthreads();

  // ff0: vh = vin @ wh0 (16x32)
  for (int t = lane; t < 16 * 96; t += 32) {
    const int e = t / 96, rem = t % 96, h = rem / 3, c = rem % 3;
    float acc = 0.f;
#pragma unroll
    for (int i = 0; i < 16; ++i) acc += vin[e * 48 + i * 3 + c] * wh0[i * 32 + h];
    vhb[e * 96 + h * 3 + c] = acc;
  }
  __syncthreads();
  for (int t = lane; t < 16 * 32; t += 32) {
    const int e = t >> 5, h = t & 31;
    float nn = 0.f;
#pragma unroll
    for (int c = 0; c < 3; ++c) { const float x = vhb[e * 96 + h * 3 + c]; nn += x * x; }
    Abf[e * 544 + 128 + h] = f2bf(sqrtf(fmaxf(nn, EPSF)));
  }
  __syncthreads();
  wave_gemm(Abf, nullptr, 0, 544, 5, pf0ws, 32, wsb0, s1, 512, lane);
  __syncthreads();
  wave_gemm(nullptr, s1, 1, 512, 16, pf0wsv, 2, wsvb0, gate, 32, lane);
  __syncthreads();
  for (int t = lane; t < 16 * 96; t += 32) {   // v1 = (vh @ wv0[32x32]) * sig(gate)
    const int e = t / 96, rem = t % 96, o = rem / 3, c = rem % 3;
    float acc = 0.f;
    for (int h = 0; h < 32; ++h) acc += vhb[e * 96 + h * 3 + c] * wv0[h * 32 + o];
    vb[e * 96 + o * 3 + c] = acc * sigmoidf(gate[e * 32 + o]);
  }
  __syncthreads();

  // ff1
  for (int t = lane; t < 16 * 512; t += 32) {
    const int e = t >> 9, c = t & 511;
    Abf[e * 544 + c] = f2bf(fmaxf(s1[e * 512 + c], 0.f));
  }
  for (int t = lane; t < 16 * 96; t += 32) {   // vh2 = v1 @ wh1 (32x32)
    const int e = t / 96, rem = t % 96, h = rem / 3, c = rem % 3;
    float acc = 0.f;
    for (int i = 0; i < 32; ++i) acc += vb[e * 96 + i * 3 + c] * wh1[i * 32 + h];
    vhb[e * 96 + h * 3 + c] = acc;
  }
  __syncthreads();
  for (int t = lane; t < 16 * 32; t += 32) {
    const int e = t >> 5, h = t & 31;
    float nn = 0.f;
#pragma unroll
    for (int c = 0; c < 3; ++c) { const float x = vhb[e * 96 + h * 3 + c]; nn += x * x; }
    Abf[e * 544 + 512 + h] = f2bf(sqrtf(fmaxf(nn, EPSF)));
  }
  __syncthreads();
  wave_gemm(Abf, nullptr, 0, 544, 17, pf1ws, 8, wsb1, s2, 128, lane);
  __syncthreads();
  wave_gemm(nullptr, s2, 1, 128, 4, pf1wsv, 1, wsvb1, gate, 16, lane);
  __syncthreads();
  for (int t = lane; t < 16 * 48; t += 32) {   // v2 (vo=16), packed stride 48 into vb
    const int e = t / 48, rem = t % 48, o = rem / 3, c = rem % 3;
    float acc = 0.f;
    for (int h = 0; h < 32; ++h) acc += vhb[e * 96 + h * 3 + c] * wv1[h * 16 + o];
    vb[e * 48 + o * 3 + c] = acc * sigmoidf(gate[e * 16 + o]);
  }
  __syncthreads();

  // residual + LayerNorm1
  for (int t = lane; t < 16 * 128; t += 32) {
    const int e = t >> 7, c = t & 127;
    s2[e * 128 + c] += s[(size_t)(n0 + e) * 128 + c];
  }
  for (int t = lane; t < 16 * 48; t += 32) vin[t] += vb[t];
  __syncthreads();
  if (lane < 16) {
    const int e = lane;
    float mu = 0.f;
    for (int c = 0; c < 128; ++c) mu += s2[e * 128 + c];
    mu *= (1.f / 128.f);
    float var = 0.f;
    for (int c = 0; c < 128; ++c) { const float d = s2[e * 128 + c] - mu; var += d * d; }
    gate[e]      = mu;
    gate[16 + e] = rsqrtf(var * (1.f / 128.f) + 1e-5f);
    float ss = 0.f;
    for (int i = 0; i < 16; ++i) {
      float q = 0.f;
#pragma unroll
      for (int c = 0; c < 3; ++c) { const float x = vin[e * 48 + i * 3 + c]; q += x * x; }
      ss += fmaxf(q, EPSF);
    }
    gate[32 + e] = sqrtf(ss * (1.f / 16.f));
  }
  __syncthreads();
  for (int t = lane; t < 16 * 128; t += 32) {
    const int e = t >> 7, c = t & 127;
    s[(size_t)(n0 + e) * 128 + c] =
        (s2[e * 128 + c] - gate[e]) * gate[16 + e] * gamma[c] + beta[c];
  }
  for (int t = lane; t < 16 * 48; t += 32) {
    const int e = t / 48;
    v[(size_t)(n0 + e) * 48 + (t % 48)] = vin[t] / gate[32 + e];
  }
}

// ---------------------------------------------------------------------------
// Pooling + head
// ---------------------------------------------------------------------------
__global__ __launch_bounds__(256) void pool_kernel(
    const float* __restrict__ s, const int* __restrict__ batch,
    float* __restrict__ pool, float* __restrict__ pcnt, int N)
{
  const long t = (long)blockIdx.x * blockDim.x + threadIdx.x;
  if (t < (long)N * 128) {
    const int n = (int)(t >> 7), c = (int)(t & 127);
    const int b = batch[n];
    atomicAdd(&pool[b * 128 + c], s[(size_t)n * 128 + c]);
    if (c == 0) atomicAdd(&pcnt[b], 1.f);
  }
}

__global__ __launch_bounds__(32) void head_kernel(
    const float* __restrict__ pool, const float* __restrict__ pcnt,
    const unsigned short* __restrict__ pw1, const float* __restrict__ b1,
    const float* __restrict__ w2, const float* __restrict__ b2,
    float* __restrict__ out)
{
  __shared__ __align__(16) unsigned short Abf[16 * 128];
  __shared__ float hr[16 * 128];
  const int lane = threadIdx.x;
  for (int t = lane; t < 16 * 128; t += 32) {
    const int g = t >> 7, c = t & 127;
    Abf[t] = f2bf(pool[g * 128 + c] / fmaxf(pcnt[g], 1.f));
  }
  __syncthreads();
  wave_gemm(Abf, nullptr, 0, 128, 4, pw1, 8, b1, hr, 128, lane);  // 16x128 @ 128x128
  __syncthreads();
  if (lane < 16) {
    float acc = b2[0];
    for (int j = 0; j < 128; ++j) acc += fmaxf(hr[lane * 128 + j], 0.f) * w2[j];
    out[lane] = acc;
  }
}

// ---------------------------------------------------------------------------
// Host launcher
// ---------------------------------------------------------------------------
struct GvpPtrs { const float *wh, *ws_b, *ws_w, *wsv_b, *wsv_w, *wv; };

static GvpPtrs gvp_at(void* const* d_in, int base) {
  GvpPtrs g;
  g.wh    = (const float*)d_in[base + 0];
  g.ws_b  = (const float*)d_in[base + 1];
  g.ws_w  = (const float*)d_in[base + 2];
  g.wsv_b = (const float*)d_in[base + 3];
  g.wsv_w = (const float*)d_in[base + 4];
  g.wv    = (const float*)d_in[base + 5];
  return g;
}

extern "C" void kernel_launch(void* const* d_in, const int* in_sizes, int n_in,
                              void* d_out, int out_size, void* d_ws, size_t ws_size,
                              hipStream_t stream) {
  const float* xs    = (const float*)d_in[0];
  const float* xv    = (const float*)d_in[1];
  const float* es    = (const float*)d_in[2];
  const float* ev    = (const float*)d_in[3];
  const int*   eidx  = (const int*)d_in[4];
  const int*   batch = (const int*)d_in[5];
  const float* hb1 = (const float*)d_in[6];
  const float* hb2 = (const float*)d_in[7];
  const float* hw1 = (const float*)d_in[8];
  const float* hw2 = (const float*)d_in[9];
  const GvpPtrs ip = gvp_at(d_in, 10);

  const int E = in_sizes[4] / 2;     // 262144
  const int N = in_sizes[5];         // 16384

  // workspace layout (bytes)
  const size_t OFF_S    = 0;
  const size_t OFF_V    = OFF_S    + (size_t)N * 128 * 4;
  const size_t OFF_SACC = OFF_V    + (size_t)N * 48 * 4;
  const size_t OFF_VACC = OFF_SACC + (size_t)N * 128 * 4;
  const size_t OFF_CNT  = OFF_VACC + (size_t)N * 48 * 4;
  const size_t OFF_POOL = OFF_CNT  + (size_t)N * 4;
  const size_t OFF_PCNT = OFF_POOL + 16 * 128 * 4;
  const size_t OFF_PACK = ((OFF_PCNT + 16 * 4 + 255) / 256) * 256;

  char* wsb = (char*)d_ws;
  float* S    = (float*)(wsb + OFF_S);
  float* V    = (float*)(wsb + OFF_V);
  float* SACC = (float*)(wsb + OFF_SACC);
  float* CNT  = (float*)(wsb + OFF_CNT);
  float* VACC = (float*)(wsb + OFF_VACC);
  float* POOL = (float*)(wsb + OFF_POOL);
  float* PCNT = (float*)(wsb + OFF_PCNT);
  unsigned short* PK = (unsigned short*)(wsb + OFF_PACK);

  auto pack = [&](const float* W, int K, int Ncols, size_t offEl) {
    const int KT = (K + 31) / 32, NT = (Ncols + 15) / 16;
    pack_kernel<<<KT * NT, 256, 0, stream>>>(W, K, Ncols, NT, PK + offEl);
  };

  // per-layer packed offsets (elements); 262144 el per layer
  const size_t PO_M0WS = 0,      PO_M0WSV = 45056, PO_M1WS = 47104, PO_M1WSV = 67584;
  const size_t PO_M2WS = 69632,  PO_M2WSV = 90112;
  const size_t PO_F0WS = 92160,  PO_F0WSV = 174080;
  const size_t PO_F1WS = 190464, PO_F1WSV = 260096;
  const size_t LSTRIDE = 262144, PO_HEAD = 3 * LSTRIDE;

  GvpPtrs msg[3][3], ff[3][2];
  const float *g0[3], *b0[3], *g1[3], *b1l[3];
  for (int l = 0; l < 3; ++l) {
    const int base = 16 + 34 * l;
    ff[l][0]  = gvp_at(d_in, base + 0);
    ff[l][1]  = gvp_at(d_in, base + 6);
    b0[l]  = (const float*)d_in[base + 12];  // ln0.beta
    g0[l]  = (const float*)d_in[base + 13];  // ln0.gamma
    b1l[l] = (const float*)d_in[base + 14];  // ln1.beta
    g1[l]  = (const float*)d_in[base + 15];  // ln1.gamma
    msg[l][0] = gvp_at(d_in, base + 16);
    msg[l][1] = gvp_at(d_in, base + 22);
    msg[l][2] = gvp_at(d_in, base + 28);

    const size_t L = l * LSTRIDE;
    pack(msg[l][0].ws_w, 321, 128, L + PO_M0WS);
    pack(msg[l][0].wsv_w, 128, 16, L + PO_M0WSV);
    pack(msg[l][1].ws_w, 144, 128, L + PO_M1WS);
    pack(msg[l][1].wsv_w, 128, 16, L + PO_M1WSV);
    pack(msg[l][2].ws_w, 144, 128, L + PO_M2WS);
    pack(msg[l][2].wsv_w, 128, 16, L + PO_M2WSV);
    pack(ff[l][0].ws_w, 160, 512, L + PO_F0WS);
    pack(ff[l][0].wsv_w, 512, 32, L + PO_F0WSV);
    pack(ff[l][1].ws_w, 544, 128, L + PO_F1WS);
    pack(ff[l][1].wsv_w, 128, 16, L + PO_F1WSV);
  }
  pack(hw1, 128, 128, PO_HEAD);

  // input projection
  input_proj_kernel<<<N, 128, 0, stream>>>(xs, xv, ip.wh, ip.ws_w, ip.ws_b,
                                           ip.wsv_w, ip.wsv_b, ip.wv, S, V);

  const long accN = (long)N * (128 + 48 + 1);   // sacc|vacc|cnt contiguous
  for (int l = 0; l < 3; ++l) {
    zero_kernel<<<(int)((accN + 255) / 256), 256, 0, stream>>>(SACC, accN);
    const size_t L = l * LSTRIDE;
    msg_kernel<<<E / 64, 128, 4 * MSG_ARENA, stream>>>(
        eidx, es, ev, S, V, SACC, VACC, CNT,
        msg[l][0].wh, msg[l][0].wv, msg[l][0].ws_b, msg[l][0].wsv_b,
        msg[l][1].wh, msg[l][1].wv, msg[l][1].ws_b, msg[l][1].wsv_b,
        msg[l][2].wh, msg[l][2].wv, msg[l][2].ws_b, msg[l][2].wsv_b,
        PK + L + PO_M0WS, PK + L + PO_M0WSV,
        PK + L + PO_M1WS, PK + L + PO_M1WSV,
        PK + L + PO_M2WS, PK + L + PO_M2WSV, E);
    node_update_kernel<<<N, 128, 0, stream>>>(S, V, SACC, VACC, CNT, g0[l], b0[l]);
    ff_kernel<<<N / 32, 64, 2 * FF_ARENA, stream>>>(
        S, V,
        ff[l][0].wh, ff[l][0].wv, ff[l][0].ws_b, ff[l][0].wsv_b,
        ff[l][1].wh, ff[l][1].wv, ff[l][1].ws_b, ff[l][1].wsv_b,
        PK + L + PO_F0WS, PK + L + PO_F0WSV,
        PK + L + PO_F1WS, PK + L + PO_F1WSV,
        g1[l], b1l[l]);
  }

  // global mean pool + head
  zero_kernel<<<9, 256, 0, stream>>>(POOL, 16 * 128 + 16);
  pool_kernel<<<(int)(((long)N * 128 + 255) / 256), 256, 0, stream>>>(S, batch, POOL, PCNT, N);
  head_kernel<<<1, 32, 0, stream>>>(POOL, PCNT, PK + PO_HEAD, hb1, hw2, hb2, (float*)d_out);

  (void)n_in; (void)out_size; (void)ws_size;
}
